// BondUpdateLayer_18373870092600
// MI455X (gfx1250) — matrix operand order repaired and checked by reading, not compiled, not run on hardware
//
#include <hip/hip_runtime.h>

typedef __attribute__((ext_vector_type(16))) _Float16 v16h;
typedef __attribute__((ext_vector_type(8)))  _Float16 v8h;
typedef __attribute__((ext_vector_type(8)))  float    v8f;
typedef __attribute__((ext_vector_type(4)))  float    v4f;

#define THREADS 256
#define WAVES   8      // wave32 on gfx1250
#define TILE_M  16     // bonds per wave-tile
#define DF      64     // feature dim
#define KIN     256    // concat input dim

union V16U { v16h v; v8h h[2]; };

__device__ __forceinline__ float softplusf(float x) {
    // stable softplus: max(x,0) + log(1 + exp(-|x|))
    // __expf/__logf lower to v_exp_f32 / v_log_f32 hardware transcendentals;
    // e in (0,1], and when e is tiny the log term vanishes against max(x,0),
    // so skipping log1p's guard-digit polynomial costs only f32 rounding noise.
    return fmaxf(x, 0.0f) + __logf(1.0f + __expf(-fabsf(x)));
}

__global__ __launch_bounds__(THREADS)
void bond_mlp_wmma(const float* __restrict__ bond_feats,
                   const float* __restrict__ atom_feats,
                   const float* __restrict__ global_feats,
                   const long long* __restrict__ atom_idx,
                   const long long* __restrict__ global_idx,
                   const float* __restrict__ W1, const float* __restrict__ B1,
                   const float* __restrict__ W2, const float* __restrict__ B2,
                   const float* __restrict__ W3, const float* __restrict__ B3,
                   float* __restrict__ out,
                   int nBond, int nTiles)
{
    // weights transposed to [N][K] f16 so B-fragments are contiguous per lane
    __shared__ _Float16 sW1T[64 * 256];              // 32 KB
    __shared__ _Float16 sW2T[64 * 64];               //  8 KB
    __shared__ _Float16 sW3T[32 * 64];               //  4 KB
    __shared__ _Float16 sStage[WAVES][TILE_M * 64];  // 16 KB, wave-private tiles

    const int t = threadIdx.x;

    // cooperative f32->f16 transpose of the (tiny, L2-resident) weights
    for (int i = t; i < 64 * 256; i += THREADS) {
        int n = i >> 8, k = i & 255;
        sW1T[i] = (_Float16)W1[k * 64 + n];
    }
    for (int i = t; i < 64 * 64; i += THREADS) {
        int n = i >> 6, k = i & 63;
        sW2T[i] = (_Float16)W2[k * 64 + n];
    }
    for (int i = t; i < 32 * 64; i += THREADS) {
        int n = i >> 6, k = i & 63;
        sW3T[i] = (_Float16)W3[k * 32 + n];
    }
    __syncthreads();

    const int wave = t >> 5;
    const int lane = t & 31;
    const int row  = lane & 15;   // A-layout: lane holds matrix row M = lane%16
    const int half = lane >> 4;   // K (A) / K-group (B) selector
    const int koff = half * 8;

    const int tile = blockIdx.x * WAVES + wave;
    if (tile >= nTiles) return;   // wave-uniform: EXEC stays all-ones inside

    const int bondBase = tile * TILE_M;
    int bond = bondBase + row;
    if (bond >= nBond) bond = nBond - 1;   // clamp gathers only; stores guarded

    // indices are streaming (read once) -> non-temporal, keep L2 for atom table
    const long long a0 = __builtin_nontemporal_load(&atom_idx[(long long)bond * 2 + 0]);
    const long long a1 = __builtin_nontemporal_load(&atom_idx[(long long)bond * 2 + 1]);
    const long long g  = __builtin_nontemporal_load(&global_idx[bond]);

    // each 32-wide K chunk of the 256-dim concat lives in exactly one source
    const float* seg[4] = {
        bond_feats   + (long long)bond * DF,   // streaming (read once)
        atom_feats   + a0 * DF,                // random gather: want L2-resident
        atom_feats   + a1 * DF,
        global_feats + g  * DF                 // small table: L2-resident
    };

    _Float16* stage = &sStage[wave][0];

    // ---------------- Layer 1: X(16x256) @ W1(256x64) + b1, softplus ------
    // A fragments: ISA 16-bit A 16x32 layout -> lane reads runs [koff,koff+8)
    // and [16+koff,16+koff+8) of row (lane%16) within each K chunk.
    v16h a1f[8];
    #pragma unroll
    for (int c = 0; c < 8; ++c) {
        const float* sp = seg[c >> 1] + (c & 1) * 32;
        v4f x0, x1, x2, x3;
        if (c < 2) {
            // bond_feats chunks: single-use stream, bypass cache retention
            x0 = __builtin_nontemporal_load((const v4f*)(sp + koff));
            x1 = __builtin_nontemporal_load((const v4f*)(sp + koff + 4));
            x2 = __builtin_nontemporal_load((const v4f*)(sp + koff + 16));
            x3 = __builtin_nontemporal_load((const v4f*)(sp + koff + 20));
        } else {
            x0 = *(const v4f*)(sp + koff);
            x1 = *(const v4f*)(sp + koff + 4);
            x2 = *(const v4f*)(sp + koff + 16);
            x3 = *(const v4f*)(sp + koff + 20);
        }
        #pragma unroll
        for (int e = 0; e < 4; ++e) {
            a1f[c][e]      = (_Float16)x0[e];
            a1f[c][e + 4]  = (_Float16)x1[e];
            a1f[c][e + 8]  = (_Float16)x2[e];
            a1f[c][e + 12] = (_Float16)x3[e];
        }
    }

    #pragma unroll
    for (int nb = 0; nb < 4; ++nb) {
        float bias = B1[nb * 16 + row];      // C/D layout: lane owns column N
        v8f acc;
        #pragma unroll
        for (int r = 0; r < 8; ++r) acc[r] = bias;
        #pragma unroll
        for (int c = 0; c < 8; ++c) {
            // B 32x16 f16 layout: lane holds col N=lane%16, K=[8*half,8*half+8) and [16+8*half,16+8*half+8)
            const _Float16* wp = &sW1T[(nb * 16 + row) * 256 + c * 32 + half * 8];
            V16U b;
            b.h[0] = *(const v8h*)(wp);
            b.h[1] = *(const v8h*)(wp + 16);
            acc = __builtin_amdgcn_wmma_f32_16x16x32_f16(
                false, a1f[c], false, b.v, (short)0, acc, false, false);
        }
        // D layout -> A layout transpose via wave-private LDS staging
        #pragma unroll
        for (int r = 0; r < 8; ++r) {
            stage[(r + 8 * half) * 64 + nb * 16 + row] = (_Float16)softplusf(acc[r]);
        }
    }
    // per-wave DS is in-order; fence compiler + drain DS counter (CDNA5 split wait)
    asm volatile("s_wait_dscnt 0x0" ::: "memory");

    // ---------------- Layer 2: H1(16x64) @ W2(64x64) + b2, softplus -------
    v16h a2f[2];
    #pragma unroll
    for (int c = 0; c < 2; ++c) {
        const _Float16* sp = &stage[row * 64 + c * 32 + koff];
        V16U u;
        u.h[0] = *(const v8h*)(sp);
        u.h[1] = *(const v8h*)(sp + 16);
        a2f[c] = u.v;
    }
    #pragma unroll
    for (int nb = 0; nb < 4; ++nb) {
        float bias = B2[nb * 16 + row];
        v8f acc;
        #pragma unroll
        for (int r = 0; r < 8; ++r) acc[r] = bias;
        #pragma unroll
        for (int c = 0; c < 2; ++c) {
            const _Float16* wp = &sW2T[(nb * 16 + row) * 64 + c * 32 + half * 8];
            V16U b;
            b.h[0] = *(const v8h*)(wp);
            b.h[1] = *(const v8h*)(wp + 16);
            acc = __builtin_amdgcn_wmma_f32_16x16x32_f16(
                false, a2f[c], false, b.v, (short)0, acc, false, false);
        }
        #pragma unroll
        for (int r = 0; r < 8; ++r) {
            stage[(r + 8 * half) * 64 + nb * 16 + row] = (_Float16)softplusf(acc[r]);
        }
    }
    asm volatile("s_wait_dscnt 0x0" ::: "memory");

    // ---------------- Layer 3: H2(16x64) @ W3(64x32) + b3 -----------------
    v16h a3f[2];
    #pragma unroll
    for (int c = 0; c < 2; ++c) {
        const _Float16* sp = &stage[row * 64 + c * 32 + koff];
        V16U u;
        u.h[0] = *(const v8h*)(sp);
        u.h[1] = *(const v8h*)(sp + 16);
        a3f[c] = u.v;
    }
    #pragma unroll
    for (int nb = 0; nb < 2; ++nb) {
        float bias = B3[nb * 16 + row];
        v8f acc;
        #pragma unroll
        for (int r = 0; r < 8; ++r) acc[r] = bias;
        #pragma unroll
        for (int c = 0; c < 2; ++c) {
            const _Float16* wp = &sW3T[(nb * 16 + row) * 64 + c * 32 + half * 8];
            V16U b;
            b.h[0] = *(const v8h*)(wp);
            b.h[1] = *(const v8h*)(wp + 16);
            acc = __builtin_amdgcn_wmma_f32_16x16x32_f16(
                false, a3f[c], false, b.v, (short)0, acc, false, false);
        }
        // D layout: lane holds out[M = r + 8*half][N = nb*16 + row], f32 out
        // output is a 128 MB write-once stream -> non-temporal stores
        #pragma unroll
        for (int r = 0; r < 8; ++r) {
            int m = bondBase + r + 8 * half;
            if (m < nBond)
                __builtin_nontemporal_store(acc[r], &out[(long long)m * 32 + nb * 16 + row]);
        }
    }
}

extern "C" void kernel_launch(void* const* d_in, const int* in_sizes, int n_in,
                              void* d_out, int out_size, void* d_ws, size_t ws_size,
                              hipStream_t stream) {
    (void)n_in; (void)out_size; (void)d_ws; (void)ws_size;
    const float*     bond_feats   = (const float*)d_in[0];
    const float*     atom_feats   = (const float*)d_in[1];
    const float*     global_feats = (const float*)d_in[2];
    const long long* atom_idx     = (const long long*)d_in[3];  // int64 per reference
    const long long* global_idx   = (const long long*)d_in[4];
    const float*     W1 = (const float*)d_in[5];
    const float*     B1 = (const float*)d_in[6];
    const float*     W2 = (const float*)d_in[7];
    const float*     B2 = (const float*)d_in[8];
    const float*     W3 = (const float*)d_in[9];
    const float*     B3 = (const float*)d_in[10];
    float* out = (float*)d_out;

    const int nBond  = in_sizes[0] / DF;                // 1,000,000
    const int nTiles = (nBond + TILE_M - 1) / TILE_M;   // 62,500
    const int blocks = (nTiles + WAVES - 1) / WAVES;    // 7,813

    bond_mlp_wmma<<<blocks, THREADS, 0, stream>>>(
        bond_feats, atom_feats, global_feats, atom_idx, global_idx,
        W1, B1, W2, B2, W3, B3, out, nBond, nTiles);
}
